// GraphSAGE_25864293056532
// MI455X (gfx1250) — matrix operand order repaired and checked by reading, not compiled, not run on hardware
//
#include <hip/hip_runtime.h>

typedef __attribute__((ext_vector_type(16))) _Float16 v16h;
typedef __attribute__((ext_vector_type(8)))  _Float16 v8h;
typedef __attribute__((ext_vector_type(8)))  float    v8f;

#define NN     10000
#define DIM    128
#define NCLS   40
#define CLSP   48
#define SCAN_T 256
#define CHUNK  ((NN + SCAN_T - 1) / SCAN_T)   // 40

// ---------------------------------------------------------------------------
// WMMA fragment loaders (layouts per CDNA5 ISA 7.12.2, 16-bit 16x16x32)
// ---------------------------------------------------------------------------
// A (16x32 f16): lane L<16 -> row M=L, K = {0..7, 16..23}; lane L+16 -> row
// M=L, K = {8..15, 24..31}.  f16 source => two contiguous 16-byte loads.
__device__ __forceinline__ v16h load_a_f16(const _Float16* __restrict__ row,
                                           int kbase, int lane) {
  const int hi8 = (lane & 16) >> 1;                 // 0 or 8
  const v8h x = *(const v8h*)(row + kbase + hi8);
  const v8h y = *(const v8h*)(row + kbase + 16 + hi8);
  v16h a;
#pragma unroll
  for (int i = 0; i < 8; ++i) { a[i] = x[i]; a[8 + i] = y[i]; }
  return a;
}

// B (32x16 f16): lane n<16 -> column N=n, K = {0..7, 16..23}; lane n+16 -> same
// columns, K = {8..15, 24..31}. W stored transposed as WT[n][k] => two 16B loads.
__device__ __forceinline__ v16h load_b_f16(const _Float16* __restrict__ WT,
                                           int ncol, int kbase, int lane) {
  const int hi8 = (lane & 16) >> 1;                 // 0 or 8
  const _Float16* p = WT + (size_t)ncol * DIM + kbase + hi8;
  const v8h x = *(const v8h*)(p);
  const v8h y = *(const v8h*)(p + 16);
  v16h b;
#pragma unroll
  for (int i = 0; i < 8; ++i) { b[i] = x[i]; b[8 + i] = y[i]; }
  return b;
}

// ---------------------------------------------------------------------------
// Weight prep: f32 -> f16 transposed; classifier padded 40 -> 48 columns.
// ---------------------------------------------------------------------------
__global__ void __launch_bounds__(256) sage_prep_kernel(
    const float* __restrict__ Ws1, const float* __restrict__ Wn1,
    const float* __restrict__ Ws2, const float* __restrict__ Wn2,
    const float* __restrict__ Wc,  const float* __restrict__ bc,
    _Float16* __restrict__ WsT1, _Float16* __restrict__ WnT1,
    _Float16* __restrict__ WsT2, _Float16* __restrict__ WnT2,
    _Float16* __restrict__ WcT,  float* __restrict__ bcp) {
  const int i = blockIdx.x * blockDim.x + threadIdx.x;
  if (i < DIM * DIM) {
    const int k = i >> 7, n = i & (DIM - 1);
    const int t = n * DIM + k;
    WsT1[t] = (_Float16)Ws1[i];
    WnT1[t] = (_Float16)Wn1[i];
    WsT2[t] = (_Float16)Ws2[i];
    WnT2[t] = (_Float16)Wn2[i];
  }
  if (i < CLSP * DIM) {
    const int n = i / DIM, k = i % DIM;             // WcT[n][k]
    WcT[i] = (n < NCLS) ? (_Float16)Wc[k * NCLS + n] : (_Float16)0.0f;
  }
  if (i < CLSP) bcp[i] = (i < NCLS) ? bc[i] : 0.0f;
}

__global__ void f32_to_f16_kernel(const float* __restrict__ in,
                                  _Float16* __restrict__ out, int n) {
  const int i = blockIdx.x * blockDim.x + threadIdx.x;
  if (i < n) out[i] = (_Float16)in[i];
}

__global__ void zero_i32_kernel(int* __restrict__ p, int n) {
  const int i = blockIdx.x * blockDim.x + threadIdx.x;
  if (i < n) p[i] = 0;
}

// ---------------------------------------------------------------------------
// CSR build: histogram -> single-block scan -> fill (integer atomics only)
// ---------------------------------------------------------------------------
__global__ void __launch_bounds__(256) sage_hist_kernel(
    const int* __restrict__ dst, int* __restrict__ cnt, int nedges) {
  const int e = blockIdx.x * blockDim.x + threadIdx.x;
  if (e < nedges) atomicAdd(&cnt[dst[e]], 1);
}

__global__ void __launch_bounds__(SCAN_T) sage_scan_kernel(
    const int* __restrict__ cnt, int* __restrict__ rowptr) {
  __shared__ int part[SCAN_T];
  const int t = threadIdx.x;
  const int base = t * CHUNK;
  int sum = 0;
  for (int i = 0; i < CHUNK; ++i) {
    const int idx = base + i;
    if (idx < NN) sum += cnt[idx];
  }
  part[t] = sum;
  __syncthreads();
  for (int ofs = 1; ofs < SCAN_T; ofs <<= 1) {   // Hillis-Steele inclusive
    const int v = (t >= ofs) ? part[t - ofs] : 0;
    __syncthreads();
    part[t] += v;
    __syncthreads();
  }
  int run = (t == 0) ? 0 : part[t - 1];          // exclusive chunk prefix
  for (int i = 0; i < CHUNK; ++i) {
    const int idx = base + i;
    if (idx < NN) { rowptr[idx] = run; run += cnt[idx]; }
  }
  if (t == SCAN_T - 1) rowptr[NN] = run;         // == E
}

__global__ void __launch_bounds__(256) sage_fill_kernel(
    const int* __restrict__ src, const int* __restrict__ dst,
    const int* __restrict__ rowptr, int* __restrict__ cur,
    int* __restrict__ csr_src, int nedges) {
  const int e = blockIdx.x * blockDim.x + threadIdx.x;
  if (e >= nedges) return;
  const int d = dst[e];
  const int pos = atomicAdd(&cur[d], 1);
  csr_src[rowptr[d] + pos] = src[e];
}

// ---------------------------------------------------------------------------
// Atomic-free mean aggregation: one wave per node, HALF-WAVE per edge.
// Each half-wave covers a full 256B row with 16 lanes x b128 loads, so one
// VMEM load instruction retires TWO edges. f32 accumulate; half-wave partials
// merged with shfl_xor(16); normalized f16 b128 store from lanes 0-15.
// ---------------------------------------------------------------------------
__global__ void __launch_bounds__(256) sage_gather_kernel(
    const _Float16* __restrict__ H, const int* __restrict__ csr,
    const int* __restrict__ rowptr, _Float16* __restrict__ Hn) {
  const int lane = threadIdx.x & 31;
  const int half = lane >> 4;                       // 0 or 1: which edge
  const int sub  = lane & 15;                       // 16B column chunk
  const int v = (int)((blockIdx.x * blockDim.x + threadIdx.x) >> 5);
  if (v >= NN) return;
  const int beg = rowptr[v], end = rowptr[v + 1];

  float a[8];
#pragma unroll
  for (int j = 0; j < 8; ++j) a[j] = 0.0f;

  int i = beg;
  for (; i + 1 < end; i += 2) {                     // 2 edges per iteration
    const int s = csr[i + half];                    // one b32 covers both
    const v8h h = *(const v8h*)(H + (size_t)s * DIM + sub * 8);
#pragma unroll
    for (int j = 0; j < 8; ++j) a[j] += (float)h[j];
  }
  if (i < end && half == 0) {                       // odd tail: half 0 only
    const int s = csr[i];
    const v8h h = *(const v8h*)(H + (size_t)s * DIM + sub * 8);
#pragma unroll
    for (int j = 0; j < 8; ++j) a[j] += (float)h[j];
  }

  // merge the two half-wave partial sums (lane-permute, no memory traffic)
#pragma unroll
  for (int j = 0; j < 8; ++j) a[j] += __shfl_xor(a[j], 16);

  if (half == 0) {
    const float sc = (end > beg) ? 1.0f / (float)(end - beg) : 0.0f;
    v8h o;
#pragma unroll
    for (int j = 0; j < 8; ++j) o[j] = (_Float16)(a[j] * sc);
    *(v8h*)(Hn + (size_t)v * DIM + sub * 8) = o;    // 16 lanes x 16B = 256B row
  }
}

// ---------------------------------------------------------------------------
// Fused SAGE layer: Out(f16) = relu(Hs @ Ws + Hn @ Wn + b)
// 1 block = 16-row node tile; wave w = 16-col tile. 8 WMMAs per wave.
// ---------------------------------------------------------------------------
__global__ void __launch_bounds__(256) sage_gemm_kernel(
    const _Float16* __restrict__ Hs, const _Float16* __restrict__ Hn,
    const _Float16* __restrict__ WsT, const _Float16* __restrict__ WnT,
    const float* __restrict__ bias, _Float16* __restrict__ Out) {
  const int lane  = threadIdx.x & 31;
  const int wave  = threadIdx.x >> 5;
  const int mbase = blockIdx.x << 4;
  const int row   = mbase + (lane & 15);
  const int ncol  = (wave << 4) + (lane & 15);
  const _Float16* srow = Hs + (size_t)row * DIM;
  const _Float16* arow = Hn + (size_t)row * DIM;
  v8f acc = {};
#pragma unroll
  for (int kk = 0; kk < DIM; kk += 32) {
    const v16h as = load_a_f16(srow, kk, lane);
    const v16h bs = load_b_f16(WsT, ncol, kk, lane);
    acc = __builtin_amdgcn_wmma_f32_16x16x32_f16(false, as, false, bs,
                                                 (short)0, acc, false, false);
    const v16h an = load_a_f16(arow, kk, lane);
    const v16h bn = load_b_f16(WnT, ncol, kk, lane);
    acc = __builtin_amdgcn_wmma_f32_16x16x32_f16(false, an, false, bn,
                                                 (short)0, acc, false, false);
  }
  // C/D layout: VGPR p -> M = p (lanes 0-15) or 8+p (lanes 16-31), N = lane&15
  const float b  = bias[ncol];
  const int   mo = mbase + ((lane & 16) >> 1);
#pragma unroll
  for (int p = 0; p < 8; ++p)
    Out[(size_t)(mo + p) * DIM + ncol] = (_Float16)fmaxf(acc[p] + b, 0.0f);
}

// ---------------------------------------------------------------------------
// Classifier: out(f32) = H @ Wc + bc  (40 cols padded to 48; 3 waves/block)
// ---------------------------------------------------------------------------
__global__ void __launch_bounds__(96) sage_cls_kernel(
    const _Float16* __restrict__ H, const _Float16* __restrict__ WcT,
    const float* __restrict__ bcp, float* __restrict__ Out) {
  const int lane  = threadIdx.x & 31;
  const int wave  = threadIdx.x >> 5;
  const int mbase = blockIdx.x << 4;
  const int row   = mbase + (lane & 15);
  const int ncol  = (wave << 4) + (lane & 15);
  const _Float16* srow = H + (size_t)row * DIM;
  v8f acc = {};
#pragma unroll
  for (int kk = 0; kk < DIM; kk += 32) {
    const v16h a = load_a_f16(srow, kk, lane);
    const v16h b = load_b_f16(WcT, ncol, kk, lane);
    acc = __builtin_amdgcn_wmma_f32_16x16x32_f16(false, a, false, b,
                                                 (short)0, acc, false, false);
  }
  const float b  = bcp[ncol];
  const int   mo = mbase + ((lane & 16) >> 1);
  if (ncol < NCLS) {                               // divergence only AFTER wmma
#pragma unroll
    for (int p = 0; p < 8; ++p)
      Out[(size_t)(mo + p) * NCLS + ncol] = acc[p] + b;
  }
}

// ---------------------------------------------------------------------------
extern "C" void kernel_launch(void* const* d_in, const int* in_sizes, int n_in,
                              void* d_out, int out_size, void* d_ws, size_t ws_size,
                              hipStream_t stream) {
  const float* features = (const float*)d_in[0];
  const int*   edge     = (const int*)d_in[1];
  const float* Ws1 = (const float*)d_in[2];
  const float* Wn1 = (const float*)d_in[3];
  const float* b1  = (const float*)d_in[4];
  const float* Ws2 = (const float*)d_in[5];
  const float* Wn2 = (const float*)d_in[6];
  const float* b2  = (const float*)d_in[7];
  const float* Wc  = (const float*)d_in[8];
  const float* bc  = (const float*)d_in[9];
  const int nedges = in_sizes[1] / 2;
  const int* src = edge;
  const int* dst = edge + nedges;
  float* out = (float*)d_out;

  // carve workspace (256B-aligned chunks)
  char* ws = (char*)d_ws;
  size_t off = 0;
  auto take = [&](size_t bytes) -> char* {
    char* p = ws + off;
    off = (off + bytes + 255) & ~(size_t)255;
    return p;
  };
  _Float16* feat16 = (_Float16*)take((size_t)NN * DIM * sizeof(_Float16));
  _Float16* hn     = (_Float16*)take((size_t)NN * DIM * sizeof(_Float16));
  _Float16* h1     = (_Float16*)take((size_t)NN * DIM * sizeof(_Float16));
  _Float16* h2     = (_Float16*)take((size_t)NN * DIM * sizeof(_Float16));
  int*      cnt    = (int*)take((size_t)NN * sizeof(int));        // also cursors
  int*      rowptr = (int*)take((size_t)(NN + 1) * sizeof(int));
  int*      csr    = (int*)take((size_t)nedges * sizeof(int));
  _Float16* WsT1   = (_Float16*)take(DIM * DIM * sizeof(_Float16));
  _Float16* WnT1   = (_Float16*)take(DIM * DIM * sizeof(_Float16));
  _Float16* WsT2   = (_Float16*)take(DIM * DIM * sizeof(_Float16));
  _Float16* WnT2   = (_Float16*)take(DIM * DIM * sizeof(_Float16));
  _Float16* WcT    = (_Float16*)take(CLSP * DIM * sizeof(_Float16));
  float*    bcp    = (float*)take(CLSP * sizeof(float));

  const int mtiles  = NN / 16;                     // 625, exact
  const int eblocks = (nedges + 255) / 256;
  const int nblocks = (NN * 32 + 255) / 256;       // 1 wave per node

  // 0) weight prep + feature f32->f16
  sage_prep_kernel<<<(DIM * DIM + 255) / 256, 256, 0, stream>>>(
      Ws1, Wn1, Ws2, Wn2, Wc, bc, WsT1, WnT1, WsT2, WnT2, WcT, bcp);
  f32_to_f16_kernel<<<(NN * DIM + 255) / 256, 256, 0, stream>>>(
      features, feat16, NN * DIM);

  // 1) CSR by destination (integer atomics only)
  zero_i32_kernel<<<(NN + 255) / 256, 256, 0, stream>>>(cnt, NN);
  sage_hist_kernel<<<eblocks, 256, 0, stream>>>(dst, cnt, nedges);
  sage_scan_kernel<<<1, SCAN_T, 0, stream>>>(cnt, rowptr);
  zero_i32_kernel<<<(NN + 255) / 256, 256, 0, stream>>>(cnt, NN);
  sage_fill_kernel<<<eblocks, 256, 0, stream>>>(src, dst, rowptr, cnt, csr,
                                                nedges);

  // 2) layer 1: gather-mean then fused WMMA GEMM
  sage_gather_kernel<<<nblocks, 256, 0, stream>>>(feat16, csr, rowptr, hn);
  sage_gemm_kernel<<<mtiles, 256, 0, stream>>>(feat16, hn, WsT1, WnT1, b1, h1);

  // 3) layer 2
  sage_gather_kernel<<<nblocks, 256, 0, stream>>>(h1, csr, rowptr, hn);
  sage_gemm_kernel<<<mtiles, 256, 0, stream>>>(h1, hn, WsT2, WnT2, b2, h2);

  // 4) classifier
  sage_cls_kernel<<<mtiles, 96, 0, stream>>>(h2, WcT, bcp, out);
}
